// Correlation_68367289417895
// MI455X (gfx1250) — matrix operand order repaired and checked by reading, not compiled, not run on hardware
//
#include <hip/hip_runtime.h>

typedef __attribute__((ext_vector_type(16))) _Float16 v16h;
typedef __attribute__((ext_vector_type(8)))  _Float16 v8h;
typedef __attribute__((ext_vector_type(4)))  _Float16 v4h;
typedef __attribute__((ext_vector_type(8)))  float    v8f;
typedef __attribute__((ext_vector_type(4)))  float    v4f;

#define NB   4      // batch
#define NC   256    // channels
#define NH   64
#define NW   96
#define ND   21     // displacements per axis
#define NOUT (ND*ND)
#define WT   32     // w-pixels per block (2 M-tiles of 16)
#define CSTR 264    // LDS row stride in halfs: 528B rows, 16B aligned
#define BJ   80     // staged B columns: j = wl + 2*dx + 20
#define OSTR 33
#define HW   (NH*NW)

__global__ __launch_bounds__(128)
void corr_wmma_kernel(const float* __restrict__ in1,
                      const float* __restrict__ in2,
                      float* __restrict__ out)
{
    __shared__ _Float16 lsA[WT * CSTR];   // in1 tile  [wl][c]  16,896 B
    __shared__ _Float16 lsB[BJ * CSTR];   // in2 window[j ][c]  42,240 B
    __shared__ float    lsO[ND * OSTR];   // out slab  [dx][wl]  2,772 B

    const int tid   = threadIdx.x;
    const int lane  = tid & 31;
    const int wave  = tid >> 5;            // 0..3
    const int mi    = wave >> 1;           // M-tile (0..1)
    const int np    = wave & 1;            // N-tile pair
    const int wbase = blockIdx.x * WT;     // 0, 32, 64
    const int hrow  = (int)blockIdx.y - 20;  // in2 source row owned by this block
    const int b     = blockIdx.z;

    // -------- zero-padded in2 rows: all paired outputs are zero --------
    if ((unsigned)hrow >= (unsigned)NH) {
        for (int dyi = 0; dyi < ND; ++dyi) {
            int h = hrow + 20 - 2 * dyi;               // in1 row for this dy
            if ((unsigned)h >= (unsigned)NH) continue;
            for (int idx = tid; idx < ND * WT; idx += 128) {
                int dxi = idx >> 5, wl = idx & (WT - 1);
                out[(((size_t)b * NOUT + dyi * ND + dxi) * NH + h) * NW
                    + wbase + wl] = 0.0f;
            }
        }
        return;
    }

    // -------- stage B ONCE: in2[b,:,hrow, wbase-20..wbase+59] -> lsB[j][c] --------
    const int jlo  = (wbase == 0) ? 20 : 0;                 // valid j window
    const int jcnt = (wbase == 0) ? 60 : ((wbase == 64) ? 52 : 80);
    {
        // zero the dy-invariant out-of-range column strips
        const int nzrow = BJ - jcnt;
        v4h z = {};
        for (int q = tid; q < nzrow * (CSTR / 4); q += 128) {
            int row  = q / (CSTR / 4);
            int col4 = q - row * (CSTR / 4);
            int j    = (row < jlo) ? row : row + jcnt;
            *(v4h*)&lsB[j * CSTR + 4 * col4] = z;
        }
        // vectorized 4x4 register-transpose copy of the valid window
        const int JQ = jcnt >> 2;
        const float* p2 = in2 + ((size_t)(b * NC) * NH + hrow) * NW
                              + (wbase + jlo - 20);
        for (int g = tid; g < 64 * JQ; g += 128) {
            int gc = g / JQ;
            int gj = g - gc * JQ;
            const float* gp = p2 + (size_t)(4 * gc) * HW + 4 * gj;
            v4f r0 = *(const v4f*)(gp);
            v4f r1 = *(const v4f*)(gp + HW);
            v4f r2 = *(const v4f*)(gp + 2 * HW);
            v4f r3 = *(const v4f*)(gp + 3 * HW);
#pragma unroll
            for (int i = 0; i < 4; ++i) {
                v4h t;
                t[0] = (_Float16)r0[i]; t[1] = (_Float16)r1[i];
                t[2] = (_Float16)r2[i]; t[3] = (_Float16)r3[i];
                *(v4h*)&lsB[(jlo + 4 * gj + i) * CSTR + 4 * gc] = t;
            }
        }
    }

    const int   laneN    = lane & 15;
    const int   k0       = (lane >> 4) << 3;   // A-frag K sub-offset: 0 or 8
    const int   kb16     = (lane >> 4) << 3;   // B-frag K sub-offset: 0 or 8
    const float inv_norm = 1.0f / (float)NC;

    for (int dyi = 0; dyi < ND; ++dyi) {
        const int h = hrow + 20 - 2 * dyi;       // in1 row pairing with hrow
        if ((unsigned)h >= (unsigned)NH) continue;   // block-uniform

        __syncthreads();   // lsA/lsO reusable; lsB staged before first use

        // ---- stage A: in1[b,:,h, wbase..wbase+31] -> lsA[wl][c] (4x4 transpose) ----
        {
            const float* p1 = in1 + ((size_t)(b * NC) * NH + h) * NW + wbase;
            for (int g = tid; g < 64 * 8; g += 128) {
                int gc = g >> 3;
                int gj = g & 7;
                const float* gp = p1 + (size_t)(4 * gc) * HW + 4 * gj;
                v4f r0 = *(const v4f*)(gp);
                v4f r1 = *(const v4f*)(gp + HW);
                v4f r2 = *(const v4f*)(gp + 2 * HW);
                v4f r3 = *(const v4f*)(gp + 3 * HW);
#pragma unroll
                for (int i = 0; i < 4; ++i) {
                    v4h t;
                    t[0] = (_Float16)r0[i]; t[1] = (_Float16)r1[i];
                    t[2] = (_Float16)r2[i]; t[3] = (_Float16)r3[i];
                    *(v4h*)&lsA[(4 * gj + i) * CSTR + 4 * gc] = t;
                }
            }
        }
        __syncthreads();

        // ---- banded GEMM: 2 independent 16x16 accumulators per wave ----
        v8f acc0 = {}; v8f acc1 = {};
        const _Float16* aRow  = &lsA[(mi * 16 + laneN) * CSTR + k0];
        const int jb0 = (mi + 2 * np) * 16 + laneN;
        const _Float16* bRow0 = &lsB[ jb0       * CSTR + kb16];
        const _Float16* bRow1 = &lsB[(jb0 + 16) * CSTR + kb16];
#pragma unroll
        for (int ks = 0; ks < NC / 32; ++ks) {
            const int kb = ks * 32;
            v8h alo = *(const v8h*)(aRow + kb);
            v8h ahi = *(const v8h*)(aRow + kb + 16);
            v16h a  = __builtin_shufflevector(alo, ahi,
                        0,1,2,3,4,5,6,7,8,9,10,11,12,13,14,15);
            v8h b0l = *(const v8h*)(bRow0 + kb);
            v8h b0h = *(const v8h*)(bRow0 + kb + 16);
            v16h f0 = __builtin_shufflevector(b0l, b0h,
                        0,1,2,3,4,5,6,7,8,9,10,11,12,13,14,15);
            v8h b1l = *(const v8h*)(bRow1 + kb);
            v8h b1h = *(const v8h*)(bRow1 + kb + 16);
            v16h f1 = __builtin_shufflevector(b1l, b1h,
                        0,1,2,3,4,5,6,7,8,9,10,11,12,13,14,15);
            acc0 = __builtin_amdgcn_wmma_f32_16x16x32_f16(
                       false, a, false, f0, (short)0, acc0, false, false);
            acc1 = __builtin_amdgcn_wmma_f32_16x16x32_f16(
                       false, a, false, f1, (short)0, acc1, false, false);
        }

        // ---- extract the even-offset band -> lsO[dx][wl] ----
#pragma unroll
        for (int r = 0; r < 8; ++r) {
            int wi = (lane < 16) ? r : (8 + r);   // D layout: M per lane-half
            int wl = mi * 16 + wi;
            {
                int j    = (mi + 2 * np) * 16 + laneN;
                int diff = j - wl - 20;           // == 2*dx when valid
                int dxi  = (diff >> 1) + 10;
                if (((diff & 1) == 0) && (unsigned)dxi < (unsigned)ND)
                    lsO[dxi * OSTR + wl] = acc0[r] * inv_norm;
            }
            {
                int j    = (mi + 2 * np + 1) * 16 + laneN;
                int diff = j - wl - 20;
                int dxi  = (diff >> 1) + 10;
                if (((diff & 1) == 0) && (unsigned)dxi < (unsigned)ND)
                    lsO[dxi * OSTR + wl] = acc1[r] * inv_norm;
            }
        }
        __syncthreads();

        // ---- coalesced store of this dy's 21 x 32 slab ----
        for (int idx = tid; idx < ND * WT; idx += 128) {
            int dxi = idx >> 5; int wl = idx & (WT - 1);
            int d   = dyi * ND + dxi;
            out[(((size_t)b * NOUT + d) * NH + h) * NW + wbase + wl] =
                lsO[dxi * OSTR + wl];
        }
    }
}

extern "C" void kernel_launch(void* const* d_in, const int* in_sizes, int n_in,
                              void* d_out, int out_size, void* d_ws, size_t ws_size,
                              hipStream_t stream) {
    (void)in_sizes; (void)n_in; (void)d_ws; (void)ws_size; (void)out_size;
    const float* in1 = (const float*)d_in[0];
    const float* in2 = (const float*)d_in[1];
    float* out = (float*)d_out;
    dim3 grid(NW / WT, NH + 2 * 20, NB);   // 3 x 104 x 4 blocks
    corr_wmma_kernel<<<grid, 128, 0, stream>>>(in1, in2, out);
}
